// MoE_75917841924151
// MI455X (gfx1250) — matrix operand order repaired and checked by reading, not compiled, not run on hardware
//
#include <hip/hip_runtime.h>
#include <hip/hip_bf16.h>
#include <math.h>
#include <stdint.h>

// ---------------------------------------------------------------------------
// MoE (top-2 gating, eval path) for MI455X / gfx1250, wave32.
//   * sparse dispatch: only B*K = 8192 (token,expert) pairs computed
//   * all GEMMs: v_wmma_f32_16x16x32_f16 (f16 operands, f32 accum)
//   * block tile 128x128, wave tile 32x64; all 6 fragments preloaded per
//     k-step, then 8 back-to-back WMMAs (no WAR hazard v_nops inside the step)
//   * tile staging: Tensor Data Mover (tensor_load_to_lds), double-buffered,
//     synchronized with s_wait_tensorcnt; LDS [rows][34] pad via D# pad fields
//   * weights pre-converted + pre-transposed to f16 once (L2-resident ~50MB)
// ---------------------------------------------------------------------------

typedef _Float16 half_t;
typedef __attribute__((ext_vector_type(16))) _Float16 v16h;
typedef __attribute__((ext_vector_type(8)))  float    v8f;
typedef __attribute__((ext_vector_type(4)))  unsigned int v4u;
typedef __attribute__((ext_vector_type(8)))  int      v8i;
typedef __attribute__((ext_vector_type(4)))  int      v4i;

#define NE   8
#define TOPK 2
#define DG   1024
#define DI   1024
#define HG   512
#define HH   2048
#define CC   512
#define BB   4096
#define NPAIR (BB * TOPK)
#define TEMP_INV (1.0f / 5.0f)
#define EPS64 2.2204460492503131e-16f

// ---------------------------------------------------------------------------
// TDM: async 2D f16 tile load (tile_rows x 32 elems) global -> LDS.
// LDS layout produced: [tile_rows][34] halfs (pad 1 dword per 16 dwords,
// i.e. D#.pad_interval=3 (16 DWORDs), D#.pad_amount=0 (1 DWORD)).
// Rows >= valid_rows read as zero (OOB vs tensor_dim1). Issued per-wave.
// ---------------------------------------------------------------------------
__device__ __forceinline__ void tdm_load_tile32(unsigned lds_addr, const void* gptr,
                                                unsigned tile_rows, unsigned valid_rows,
                                                unsigned stride_elems) {
  unsigned long long ga = (unsigned long long)(uintptr_t)gptr;
  v4u g0;
  g0.x = 1u;                                     // count=1, user descriptor
  g0.y = lds_addr;                               // LDS byte address
  g0.z = (unsigned)ga;                           // global_addr[31:0]
  g0.w = (unsigned)(ga >> 32) | (2u << 30);      // global_addr[56:32] | type=2
  v8i g1;
  g1[0] = (int)((1u << 16) |                     // data_size = 1 (2 bytes)
                (1u << 20) |                     // pad_enable
                (3u << 22));                     // pad_interval: 16 DWORDs
                                                 // pad_amount = 0 -> 1 DWORD
  g1[1] = (int)(32u << 16);                      // tensor_dim0[15:0] = 32
  g1[2] = (int)((valid_rows & 0xFFFFu) << 16);   // tensor_dim1[15:0]
  g1[3] = (int)(32u << 16);                      // tile_dim0 = 32
  g1[4] = (int)(tile_rows & 0xFFFFu);            // tile_dim1 (tile_dim2=0)
  g1[5] = (int)stride_elems;                     // tensor_dim0_stride[31:0]
  g1[6] = 0;
  g1[7] = 0;
  v4i z4 = {0, 0, 0, 0};                         // 2D: groups 2/3 unused
  v8i z8 = {0, 0, 0, 0, 0, 0, 0, 0};
  __builtin_amdgcn_tensor_load_to_lds(g0, g1, z4, z4, z8, 0);
}

__device__ __forceinline__ unsigned lds_off(const void* p) {
  return (unsigned)(uintptr_t)p;                 // low 32 bits of LDS aperture addr
}

// ---------------------------------------------------------------------------
// small utility kernels
// ---------------------------------------------------------------------------
__global__ void moe_zero_i32(int* p, int n) {
  int i = blockIdx.x * blockDim.x + threadIdx.x;
  if (i < n) p[i] = 0;
}

// f32 -> f16 elementwise
__global__ void moe_cvt_f16(const float* __restrict__ src, half_t* __restrict__ dst, int n) {
  int i = blockIdx.x * blockDim.x + threadIdx.x;
  if (i < n) dst[i] = (half_t)src[i];
}

// batched transpose + convert: dst[b][c][r] (f16) = src[b][r][c] (f32)
__global__ __launch_bounds__(256) void moe_transpose_cvt(
    const float* __restrict__ src, half_t* __restrict__ dst,
    int R, int C, size_t sBatch, size_t dBatch) {
  __shared__ float t[32][33];
  src += (size_t)blockIdx.z * sBatch;
  dst += (size_t)blockIdx.z * dBatch;
  const int c0 = blockIdx.x * 32, r0 = blockIdx.y * 32;
  const int tc = threadIdx.x & 31, tr = threadIdx.x >> 5;   // 32 x 8
  #pragma unroll
  for (int i = 0; i < 4; ++i)
    t[tr + 8 * i][tc] = src[(size_t)(r0 + tr + 8 * i) * C + c0 + tc];
  __syncthreads();
  #pragma unroll
  for (int i = 0; i < 4; ++i)
    dst[(size_t)(c0 + tr + 8 * i) * R + r0 + tc] = (half_t)t[tc][tr + 8 * i];
}

// logits = Ag[B,HG] @ Wg2[HG,E] + bg2   (tiny N=8, plain VALU)
__global__ void moe_gating2(const float* __restrict__ Ag, const float* __restrict__ Wg2,
                            const float* __restrict__ bg2, float* __restrict__ logits) {
  int idx = blockIdx.x * blockDim.x + threadIdx.x;
  if (idx >= BB * NE) return;
  int b = idx >> 3, e = idx & 7;
  const float* a = Ag + (size_t)b * HG;
  float s = bg2[e];
  #pragma unroll 4
  for (int i = 0; i < HG; ++i) s += a[i] * Wg2[i * NE + e];
  logits[idx] = s;
}

__global__ void moe_top2(const float* __restrict__ logits,
                         int* __restrict__ topIdx, float* __restrict__ topGate,
                         int* __restrict__ counts, float* __restrict__ importance,
                         float* __restrict__ loadv) {
  int b = blockIdx.x * blockDim.x + threadIdx.x;
  if (b >= BB) return;
  const float* l = logits + (size_t)b * NE;
  float v0 = -INFINITY, v1 = -INFINITY; int i0 = 0, i1 = 0;
  #pragma unroll
  for (int e = 0; e < NE; ++e) {
    float v = l[e];
    if (v > v0) { v1 = v0; i1 = i0; v0 = v; i0 = e; }
    else if (v > v1) { v1 = v; i1 = e; }
  }
  float a0 = v0 * TEMP_INV, a1 = v1 * TEMP_INV;
  float m  = fmaxf(a0, a1);
  float e0 = __expf(a0 - m), e1 = __expf(a1 - m);
  float inv = 1.0f / (e0 + e1);
  float g0 = e0 * inv, g1 = e1 * inv;
  topIdx[2 * b] = i0;   topIdx[2 * b + 1] = i1;
  topGate[2 * b] = g0;  topGate[2 * b + 1] = g1;
  atomicAdd(&counts[i0], 1);       atomicAdd(&counts[i1], 1);
  atomicAdd(&importance[i0], g0);  atomicAdd(&importance[i1], g1);
  atomicAdd(&loadv[i0], 1.0f);     atomicAdd(&loadv[i1], 1.0f);
}

__global__ void moe_offsets(const int* __restrict__ counts, int* __restrict__ offsets) {
  if (threadIdx.x == 0 && blockIdx.x == 0) {
    int acc = 0;
    #pragma unroll
    for (int e = 0; e < NE; ++e) { offsets[e] = acc; acc += counts[e]; }
  }
}

__global__ void moe_scatter(const int* __restrict__ topIdx, const int* __restrict__ offsets,
                            int* __restrict__ cursor, int* __restrict__ pairTok,
                            int* __restrict__ pairSlot) {
  int idx = blockIdx.x * blockDim.x + threadIdx.x;
  if (idx >= NPAIR) return;
  int b = idx >> 1;
  int e = topIdx[idx];
  int pos = offsets[e] + atomicAdd(&cursor[e], 1);
  pairTok[pos] = b;
  pairSlot[idx] = pos;
}

// pre-gather dispatched rows: Xg[p,:] = (f16) x2[pairTok[p],:]
__global__ void moe_gather(const float* __restrict__ x2, const int* __restrict__ pairTok,
                           half_t* __restrict__ Xg) {
  int idx = blockIdx.x * blockDim.x + threadIdx.x;   // NPAIR * DI/8
  if (idx >= NPAIR * (DI / 8)) return;
  int row = idx >> 7;                                // DI/8 = 128
  int c = (idx & 127) * 8;
  const float* s = x2 + (size_t)pairTok[row] * DI + c;
  half_t* d = Xg + (size_t)row * DI + c;
  #pragma unroll
  for (int i = 0; i < 8; ++i) d[i] = (half_t)s[i];
}

// ---------------------------------------------------------------------------
// WMMA GEMM core (shared by the 3 GEMMs):
//   block = 256 thr (8 waves), out tile 128 x 128, K-step 32
//   wave tile 32 x 64: preload af[2] + bf[4], then 8 back-to-back WMMAs
//   A: f16 row-major [rows][lda], B: f16 pre-transposed [N][ldb] (row = col)
// ---------------------------------------------------------------------------
#define GEMM_CORE(NITER, ABASE, LDA, VALIDA, BBASE, LDB)                          \
  __shared__ half_t As[2][128][34];                                               \
  __shared__ half_t Bs[2][128][34];                                               \
  const int tid = threadIdx.x, wave = tid >> 5, lane = tid & 31;                  \
  const int wm = wave & 3, wn = wave >> 2;                                        \
  const int laneN = lane & 15, hi = lane >> 4;                                    \
  v8f acc[2][4] = {};                                                             \
  if (wave == 0) {                                                                \
    tdm_load_tile32(lds_off(&As[0][0][0]), (ABASE), 128, (VALIDA), (LDA));        \
    tdm_load_tile32(lds_off(&Bs[0][0][0]), (BBASE), 128, 128, (LDB));             \
  }                                                                               \
  for (int it = 0; it < (NITER); ++it) {                                          \
    __syncthreads();                                                              \
    if (wave == 0) {                                                              \
      if (it + 1 < (NITER)) {                                                     \
        const int nb = (it + 1) & 1, k1 = (it + 1) * 32;                          \
        tdm_load_tile32(lds_off(&As[nb][0][0]), (ABASE) + k1, 128, (VALIDA), (LDA)); \
        tdm_load_tile32(lds_off(&Bs[nb][0][0]), (BBASE) + k1, 128, 128, (LDB));   \
        __builtin_amdgcn_s_wait_tensorcnt(2);                                     \
      } else {                                                                    \
        __builtin_amdgcn_s_wait_tensorcnt(0);                                     \
      }                                                                           \
    }                                                                             \
    __syncthreads();                                                              \
    const int buf = it & 1;                                                       \
    v16h af[2], bf[4];                                                            \
    _Pragma("unroll")                                                             \
    for (int i = 0; i < 2; ++i) {                                                 \
      const int m = wm * 32 + i * 16 + laneN;                                     \
      _Pragma("unroll")                                                           \
      for (int q = 0; q < 16; ++q)                                                \
        af[i][q] = As[buf][m][(q & 7) + ((q >> 3) << 4) + (hi << 3)];             \
    }                                                                             \
    _Pragma("unroll")                                                             \
    for (int j = 0; j < 4; ++j) {                                                 \
      const int n = wn * 64 + j * 16 + laneN;                                     \
      _Pragma("unroll")                                                           \
      for (int q = 0; q < 16; ++q) bf[j][q] = Bs[buf][n][(q & 7) + ((q >> 3) << 4) + (hi << 3)]; \
    }                                                                             \
    _Pragma("unroll")                                                             \
    for (int j = 0; j < 4; ++j) {                                                 \
      acc[0][j] = __builtin_amdgcn_wmma_f32_16x16x32_f16(false, af[0], false, bf[j], \
                                                  (short)0, acc[0][j], false, false); \
      acc[1][j] = __builtin_amdgcn_wmma_f32_16x16x32_f16(false, af[1], false, bf[j], \
                                                  (short)0, acc[1][j], false, false); \
    }                                                                             \
  }

// GEMM #1: Ag = relu(xh @ Wg1t^T + bg1), dense [4096,1024]x[1024,512]
__global__ __launch_bounds__(256) void moe_gating1_wmma(
    const half_t* __restrict__ xh, const half_t* __restrict__ Wg1t,
    const float* __restrict__ bg1, float* __restrict__ Ag) {
  const int rowBase = blockIdx.y * 128, colBase = blockIdx.x * 128;
  const half_t* Abase = xh + (size_t)rowBase * DG;
  const half_t* Bbase = Wg1t + (size_t)colBase * DG;
  GEMM_CORE(DG / 32, Abase, DG, 128, Bbase, DG)
  #pragma unroll
  for (int i = 0; i < 2; ++i) {
    #pragma unroll
    for (int j = 0; j < 4; ++j) {
      const int n = colBase + wn * 64 + j * 16 + laneN;
      const float bv = bg1[n];
      #pragma unroll
      for (int v = 0; v < 8; ++v) {
        const int mrow = rowBase + wm * 32 + i * 16 + v + (hi << 3);
        float val = acc[i][j][v] + bv;
        Ag[(size_t)mrow * HG + n] = val > 0.f ? val : 0.f;
      }
    }
  }
}

// GEMM #2: per-expert h = relu(Xg_seg @ W1t[e]^T + b1[e]) -> f16 Hbuf
__global__ __launch_bounds__(256) void moe_expert_ffn1_wmma(
    const half_t* __restrict__ Xg, const half_t* __restrict__ W1t,
    const float* __restrict__ b1, const int* __restrict__ counts,
    const int* __restrict__ offsets, half_t* __restrict__ Hbuf) {
  const int e = blockIdx.z;
  const int cnt = counts[e];
  const int rowBase = blockIdx.y * 128;
  if (rowBase >= cnt) return;
  const int off = offsets[e];
  const int colBase = blockIdx.x * 128;
  const int validA = cnt - rowBase;                 // TDM zero-fills rows >= validA
  const half_t* Abase = Xg + (size_t)(off + rowBase) * DI;
  const half_t* Bbase = W1t + ((size_t)e * HH + colBase) * DI;
  GEMM_CORE(DI / 32, Abase, DI, validA, Bbase, DI)
  #pragma unroll
  for (int i = 0; i < 2; ++i) {
    #pragma unroll
    for (int j = 0; j < 4; ++j) {
      const int n = colBase + wn * 64 + j * 16 + laneN;
      const float bv = b1[(size_t)e * HH + n];
      #pragma unroll
      for (int v = 0; v < 8; ++v) {
        const int r = rowBase + wm * 32 + i * 16 + v + (hi << 3);
        if (r < cnt) {
          float val = acc[i][j][v] + bv;
          Hbuf[(size_t)(off + r) * HH + n] = (half_t)(val > 0.f ? val : 0.f);
        }
      }
    }
  }
}

// GEMM #3: per-expert out = h_seg @ W2t[e]^T + b2[e] -> f32 Outp
__global__ __launch_bounds__(256) void moe_expert_ffn2_wmma(
    const half_t* __restrict__ Hbuf, const half_t* __restrict__ W2t,
    const float* __restrict__ b2, const int* __restrict__ counts,
    const int* __restrict__ offsets, float* __restrict__ Outp) {
  const int e = blockIdx.z;
  const int cnt = counts[e];
  const int rowBase = blockIdx.y * 128;
  if (rowBase >= cnt) return;
  const int off = offsets[e];
  const int colBase = blockIdx.x * 128;
  const int validA = cnt - rowBase;
  const half_t* Abase = Hbuf + (size_t)(off + rowBase) * HH;
  const half_t* Bbase = W2t + ((size_t)e * CC + colBase) * HH;
  GEMM_CORE(HH / 32, Abase, HH, validA, Bbase, HH)
  #pragma unroll
  for (int i = 0; i < 2; ++i) {
    #pragma unroll
    for (int j = 0; j < 4; ++j) {
      const int n = colBase + wn * 64 + j * 16 + laneN;
      const float bv = b2[(size_t)e * CC + n];
      #pragma unroll
      for (int v = 0; v < 8; ++v) {
        const int r = rowBase + wm * 32 + i * 16 + v + (hi << 3);
        if (r < cnt)
          Outp[(size_t)(off + r) * CC + n] = acc[i][j][v] + bv;
      }
    }
  }
}

// ---------------------------------------------------------------------------
// combine: per token, softmax over its 2 pair rows, gate-weighted sum, log
// ---------------------------------------------------------------------------
__global__ __launch_bounds__(256) void moe_combine(
    const float* __restrict__ Outp, const int* __restrict__ pairSlot,
    const float* __restrict__ topGate, float* __restrict__ out) {
  const int b = blockIdx.x, tid = threadIdx.x;
  __shared__ float r0[CC], r1[CC], red[256];
  const int p0 = pairSlot[2 * b], p1 = pairSlot[2 * b + 1];
  const float g0 = topGate[2 * b], g1 = topGate[2 * b + 1];
  for (int c = tid; c < CC; c += 256) {
    r0[c] = Outp[(size_t)p0 * CC + c];
    r1[c] = Outp[(size_t)p1 * CC + c];
  }
  __syncthreads();
  float a0 = fmaxf(r0[tid], r0[tid + 256]);
  float a1 = fmaxf(r1[tid], r1[tid + 256]);
  red[tid] = a0; __syncthreads();
  for (int s = 128; s > 0; s >>= 1) { if (tid < s) red[tid] = fmaxf(red[tid], red[tid + s]); __syncthreads(); }
  const float m0 = red[0]; __syncthreads();
  red[tid] = a1; __syncthreads();
  for (int s = 128; s > 0; s >>= 1) { if (tid < s) red[tid] = fmaxf(red[tid], red[tid + s]); __syncthreads(); }
  const float m1 = red[0]; __syncthreads();
  red[tid] = __expf(r0[tid] - m0) + __expf(r0[tid + 256] - m0); __syncthreads();
  for (int s = 128; s > 0; s >>= 1) { if (tid < s) red[tid] += red[tid + s]; __syncthreads(); }
  const float S0 = red[0]; __syncthreads();
  red[tid] = __expf(r1[tid] - m1) + __expf(r1[tid + 256] - m1); __syncthreads();
  for (int s = 128; s > 0; s >>= 1) { if (tid < s) red[tid] += red[tid + s]; __syncthreads(); }
  const float S1 = red[0]; __syncthreads();
  const float c0 = g0 / S0, c1 = g1 / S1;
  for (int c = tid; c < CC; c += 256) {
    float comb = c0 * __expf(r0[c] - m0) + c1 * __expf(r1[c] - m1);
    if (comb == 0.0f) comb = EPS64;
    out[(size_t)b * CC + c] = __logf(comb);
  }
}

// cv_squared(importance) + cv_squared(load), unbiased var (ddof=1)
__global__ void moe_loss(const float* __restrict__ importance,
                         const float* __restrict__ loadv, float* __restrict__ out_loss) {
  if (threadIdx.x != 0 || blockIdx.x != 0) return;
  float mi = 0.f, ml = 0.f;
  #pragma unroll
  for (int e = 0; e < NE; ++e) { mi += importance[e]; ml += loadv[e]; }
  mi *= (1.0f / NE); ml *= (1.0f / NE);
  float vi = 0.f, vl = 0.f;
  #pragma unroll
  for (int e = 0; e < NE; ++e) {
    float di = importance[e] - mi; vi += di * di;
    float dl = loadv[e] - ml;      vl += dl * dl;
  }
  vi *= (1.0f / (NE - 1)); vl *= (1.0f / (NE - 1));
  *out_loss = 0.01f * (vi / (mi * mi + 1e-10f) + vl / (ml * ml + 1e-10f));
}

// ---------------------------------------------------------------------------
// launcher
// ---------------------------------------------------------------------------
extern "C" void kernel_launch(void* const* d_in, const int* in_sizes, int n_in,
                              void* d_out, int out_size, void* d_ws, size_t ws_size,
                              hipStream_t stream) {
  const float* x   = (const float*)d_in[0];
  const float* x2  = (const float*)d_in[1];
  const float* Wg1 = (const float*)d_in[2];
  const float* bg1 = (const float*)d_in[3];
  const float* Wg2 = (const float*)d_in[4];
  const float* bg2 = (const float*)d_in[5];
  const float* W1  = (const float*)d_in[6];
  const float* b1  = (const float*)d_in[7];
  const float* W2  = (const float*)d_in[8];
  const float* b2  = (const float*)d_in[9];
  float* out = (float*)d_out;   // [B*C] log-probs, then [1] loss

  char* w = (char*)d_ws;
  half_t* xh   = (half_t*)w;  w += (size_t)BB * DG * sizeof(half_t);        // 8 MB
  half_t* Wg1t = (half_t*)w;  w += (size_t)HG * DG * sizeof(half_t);        // 1 MB
  half_t* W1t  = (half_t*)w;  w += (size_t)NE * HH * DI * sizeof(half_t);   // 32 MB
  half_t* W2t  = (half_t*)w;  w += (size_t)NE * CC * HH * sizeof(half_t);   // 16 MB
  half_t* Xg   = (half_t*)w;  w += (size_t)NPAIR * DI * sizeof(half_t);     // 16 MB
  half_t* Hbuf = (half_t*)w;  w += (size_t)NPAIR * HH * sizeof(half_t);     // 32 MB
  float*  Ag   = (float*)w;   w += (size_t)BB * HG * sizeof(float);         // 8 MB
  float*  Outp = (float*)w;   w += (size_t)NPAIR * CC * sizeof(float);      // 16 MB
  float*  logits  = (float*)w; w += (size_t)BB * NE * sizeof(float);
  int*    topIdx  = (int*)w;   w += (size_t)BB * TOPK * sizeof(int);
  float*  topGate = (float*)w; w += (size_t)BB * TOPK * sizeof(float);
  int*    pairTok = (int*)w;   w += (size_t)NPAIR * sizeof(int);
  int*    pairSlot= (int*)w;   w += (size_t)NPAIR * sizeof(int);
  int*    counts  = (int*)w;   w += NE * sizeof(int);
  int*    offsets = (int*)w;   w += NE * sizeof(int);
  int*    cursor  = (int*)w;   w += NE * sizeof(int);
  float*  importance = (float*)w; w += NE * sizeof(float);
  float*  loadv   = (float*)w; w += NE * sizeof(float);

  // 1. zero counters (counts..loadv contiguous, 5*NE words)
  moe_zero_i32<<<1, 64, 0, stream>>>(counts, 5 * NE);
  // 2. f16 conversions / transposes (one-time pre-pass, ~50MB L2-resident)
  moe_cvt_f16<<<(BB * DG + 255) / 256, 256, 0, stream>>>(x, xh, BB * DG);
  moe_transpose_cvt<<<dim3(HG / 32, DG / 32, 1), 256, 0, stream>>>(
      Wg1, Wg1t, DG, HG, 0, 0);
  moe_transpose_cvt<<<dim3(HH / 32, DI / 32, NE), 256, 0, stream>>>(
      W1, W1t, DI, HH, (size_t)DI * HH, (size_t)HH * DI);
  moe_transpose_cvt<<<dim3(CC / 32, HH / 32, NE), 256, 0, stream>>>(
      W2, W2t, HH, CC, (size_t)HH * CC, (size_t)CC * HH);
  // 3. gating layer 1 (WMMA + TDM)
  moe_gating1_wmma<<<dim3(HG / 128, BB / 128), 256, 0, stream>>>(xh, Wg1t, bg1, Ag);
  // 4. gating layer 2
  moe_gating2<<<(BB * NE + 255) / 256, 256, 0, stream>>>(Ag, Wg2, bg2, logits);
  // 5. top-2 + gates + stats
  moe_top2<<<(BB + 255) / 256, 256, 0, stream>>>(logits, topIdx, topGate,
                                                 counts, importance, loadv);
  // 6. prefix over experts
  moe_offsets<<<1, 1, 0, stream>>>(counts, offsets);
  // 7. scatter pairs into per-expert segments
  moe_scatter<<<(NPAIR + 255) / 256, 256, 0, stream>>>(topIdx, offsets, cursor,
                                                       pairTok, pairSlot);
  // 8. pre-gather dispatched activation rows (f16)
  moe_gather<<<(NPAIR * (DI / 8) + 255) / 256, 256, 0, stream>>>(x2, pairTok, Xg);
  // 9. expert layer 1 (WMMA + TDM), worst-case row tiles, early-exit on counts
  moe_expert_ffn1_wmma<<<dim3(HH / 128, BB / 128, NE), 256, 0, stream>>>(
      Xg, W1t, b1, counts, offsets, Hbuf);
  // 10. expert layer 2 (WMMA + TDM)
  moe_expert_ffn2_wmma<<<dim3(CC / 128, BB / 128, NE), 256, 0, stream>>>(
      Hbuf, W2t, b2, counts, offsets, Outp);
  // 11. softmax + gated combine + log
  moe_combine<<<BB, 256, 0, stream>>>(Outp, pairSlot, topGate, out);
  // 12. aux loss scalar
  moe_loss<<<1, 32, 0, stream>>>(importance, loadv, out + (size_t)BB * CC);
}
